// RepformerLayer_57062935494719
// MI455X (gfx1250) — matrix operand order repaired and checked by reading, not compiled, not run on hardware
//
#include <hip/hip_runtime.h>
#include <hip/hip_bf16.h>
#include <math.h>

// ---------------- problem constants ----------------
#define PNB   4
#define PNLOC 512
#define PNALL 768
#define PNNEI 64
#define PNG1  128
#define PNG2  32
#define PAXIS 4
#define PND2  32
#define PNH2  4
#define PND1  64
#define PNH1  4
#define PG1IN 800
#define PSHIFT 20.0f

typedef __bf16 v16bf __attribute__((ext_vector_type(16)));
typedef float  v8f   __attribute__((ext_vector_type(8)));

// f32 -> bf16 round-to-nearest-even
static __device__ inline __bf16 f2bf(float x) {
  unsigned u = __builtin_bit_cast(unsigned, x);
  unsigned r = (u + 0x7fffu + ((u >> 16) & 1u)) >> 16;
  unsigned short s = (unsigned short)r;
  return __builtin_bit_cast(__bf16, s);
}

// ---------------- WMMA fragment helpers (gfx1250, wave32) ----------------
// A fragment: 16x32 bf16 tile (row-major, stride ld). contiguous k order used here (B helpers match):
// lanes 0-15: m=lane, K 0..15 across vgprs; lanes 16-31: K 16..31.
static __device__ inline v16bf a_frag_bf(const __bf16* A, int ld) {
  int l = threadIdx.x & 31;
  const __bf16* row = A + (l & 15) * ld;
  int kb = (l >> 4) << 4;
  v16bf a;
#pragma unroll
  for (int p = 0; p < 8; ++p) {
    int k = kb + 2 * p;
    a[2 * p]     = row[k];
    a[2 * p + 1] = row[k + 1];
  }
  return a;
}
// Same, converting from f32 source on the fly.
static __device__ inline v16bf a_frag_f32(const float* A, int ld) {
  int l = threadIdx.x & 31;
  const float* row = A + (l & 15) * ld;
  int kb = (l >> 4) << 4;
  v16bf a;
#pragma unroll
  for (int p = 0; p < 8; ++p) {
    int k = kb + 2 * p;
    a[2 * p]     = f2bf(row[k]);
    a[2 * p + 1] = f2bf(row[k + 1]);
  }
  return a;
}
// B fragment: 32x16 tile from f32 row-major source, stride ld.
// lanes 0-15: n=lane, K 0..15 across vgprs; lanes 16-31: K 16..31.
static __device__ inline v16bf b_frag_f32(const float* B, int ld) {
  int l = threadIdx.x & 31;
  int n = l & 15;
  int kb = (l >> 4) << 4;
  v16bf b;
#pragma unroll
  for (int p = 0; p < 8; ++p) {
    b[2 * p]     = f2bf(B[(size_t)(kb + 2 * p) * ld + n]);
    b[2 * p + 1] = f2bf(B[(size_t)(kb + 2 * p + 1) * ld + n]);
  }
  return b;
}
// B fragment from f32 source with a column stride (strided head slicing):
// B[k][n] = S[k*ld + n*cs]
static __device__ inline v16bf b_frag_f32_cs(const float* S, int ld, int cs) {
  int l = threadIdx.x & 31;
  int n = (l & 15) * cs;
  int kb = (l >> 4) << 4;
  v16bf b;
#pragma unroll
  for (int p = 0; p < 8; ++p) {
    b[2 * p]     = f2bf(S[(size_t)(kb + 2 * p) * ld + n]);
    b[2 * p + 1] = f2bf(S[(size_t)(kb + 2 * p + 1) * ld + n]);
  }
  return b;
}
// B fragment where logical B[k][n] = S[n*ld + k] (transposed bf16 source) — for q·k^T.
static __device__ inline v16bf b_frag_trans_bf(const __bf16* S, int ld) {
  int l = threadIdx.x & 31;
  const __bf16* col = S + (l & 15) * ld;
  int kb = (l >> 4) << 4;
  v16bf b;
#pragma unroll
  for (int p = 0; p < 8; ++p) {
    b[2 * p]     = col[kb + 2 * p];
    b[2 * p + 1] = col[kb + 2 * p + 1];
  }
  return b;
}
// C/D store: f32 16x16, M = reg + 8*(lane>=16), N = lane&15
static __device__ inline void store_c(float* C, int ld, const v8f& c) {
  int l = threadIdx.x & 31;
  int n = l & 15;
  int mb = (l >> 4) << 3;
#pragma unroll
  for (int r = 0; r < 8; ++r) C[(size_t)(mb + r) * ld + n] = c[r];
}

static __device__ inline v8f wmma_bf16(v16bf a, v16bf b, v8f c) {
  return __builtin_amdgcn_wmma_f32_16x16x32_bf16(false, a, false, b, (short)0, c,
                                                 false, false);
}

// ---------------- K1: neighbor gather gg1 = g1_ext[nlist] ----------------
__global__ void __launch_bounds__(256) k1_gather(const float* __restrict__ g1_ext,
                                                 const int* __restrict__ nlist,
                                                 float* __restrict__ gg1) {
  size_t i = (size_t)blockIdx.x * 256 + threadIdx.x;  // < NB*NLOC*NNEI*NG1
  int x = (int)(i & (PNG1 - 1));
  size_t r = i >> 7;               // (b,l,n) flat
  int b = (int)(r >> 15);          // NLOC*NNEI = 32768
  int t = nlist[r];
  gg1[i] = g1_ext[((size_t)b * PNALL + t) * PNG1 + x];
}

// ---------------- K2: fused g2 path, one workgroup per (b,l) ----------------
// computes g2_new = 0.5*(g2 + tanh(g2@W2+b) + gg@Wg1g1 + LN(attn(g2)))
constexpr size_t K2_OFF_G2   = 0;                          // bf16 64x32
constexpr size_t K2_OFF_GG   = K2_OFF_G2 + 64 * 32 * 2;    // bf16 64x128
constexpr size_t K2_OFF_QK   = K2_OFF_GG + 64 * 128 * 2;   // f32 64x256
constexpr size_t K2_OFF_V    = K2_OFF_QK + 64 * 256 * 4;   // f32 64x128
constexpr size_t K2_OFF_HHT  = K2_OFF_V + 64 * 128 * 4;    // f32 64x64
constexpr size_t K2_OFF_ATT  = K2_OFF_HHT + 64 * 64 * 4;   // f32 64x64
constexpr size_t K2_OFF_QH   = K2_OFF_ATT + 64 * 64 * 4;   // bf16 64x32
constexpr size_t K2_OFF_KH   = K2_OFF_QH + 64 * 32 * 2;    // bf16 64x32
constexpr size_t K2_OFF_AA   = K2_OFF_KH + 64 * 32 * 2;    // bf16 64x64
constexpr size_t K2_OFF_MH   = K2_OFF_AA + 64 * 64 * 2;    // f32 64x128
constexpr size_t K2_OFF_ACC  = K2_OFF_MH + 64 * 128 * 4;   // f32 64x32
constexpr size_t K2_OFF_G1   = K2_OFF_ACC + 64 * 32 * 4;   // f32 128
constexpr size_t K2_OFF_SW   = K2_OFF_G1 + 128 * 4;        // f32 64
constexpr size_t K2_OFF_H2   = K2_OFF_SW + 64 * 4;         // f32 64x4 (padded 3->4)
constexpr size_t K2_OFF_RMAX = K2_OFF_H2 + 64 * 4 * 4;     // f32 64
constexpr size_t K2_OFF_RSUM = K2_OFF_RMAX + 64 * 4;       // f32 64
constexpr size_t K2_SMEM     = K2_OFF_RSUM + 64 * 4;       // ~206 KB (<320KB WGP LDS)

__global__ void __launch_bounds__(256) k2_g2_fused(
    const float* __restrict__ g2, const float* __restrict__ gg1,
    const float* __restrict__ g1_ext, const float* __restrict__ sw,
    const float* __restrict__ h2, const float* __restrict__ w_lin2,
    const float* __restrict__ b_lin2, const float* __restrict__ w_g1g1g2,
    const float* __restrict__ w_qk2, const float* __restrict__ w_v2,
    const float* __restrict__ w_hm2, const float* __restrict__ b_hm2,
    const float* __restrict__ ln_g, const float* __restrict__ ln_b,
    float* __restrict__ g2_new) {
  extern __shared__ char smem[];
  __bf16* s_g2  = (__bf16*)(smem + K2_OFF_G2);
  __bf16* s_gg  = (__bf16*)(smem + K2_OFF_GG);
  float*  s_qk  = (float*)(smem + K2_OFF_QK);
  float*  s_v   = (float*)(smem + K2_OFF_V);
  float*  s_hht = (float*)(smem + K2_OFF_HHT);
  float*  s_att = (float*)(smem + K2_OFF_ATT);
  __bf16* s_qh  = (__bf16*)(smem + K2_OFF_QH);
  __bf16* s_kh  = (__bf16*)(smem + K2_OFF_KH);
  __bf16* s_aa  = (__bf16*)(smem + K2_OFF_AA);
  float*  s_mh  = (float*)(smem + K2_OFF_MH);
  float*  s_acc = (float*)(smem + K2_OFF_ACC);
  float*  s_g1  = (float*)(smem + K2_OFF_G1);
  float*  s_sw  = (float*)(smem + K2_OFF_SW);
  float*  s_h2  = (float*)(smem + K2_OFF_H2);
  float*  s_rmax = (float*)(smem + K2_OFF_RMAX);
  float*  s_rsum = (float*)(smem + K2_OFF_RSUM);

  const int tid = threadIdx.x;
  const int wave = tid >> 5;
  const int bl = blockIdx.x;
  const int b = bl >> 9, l = bl & (PNLOC - 1);
  const float* g2row  = g2 + (size_t)bl * PNNEI * PNG2;
  const float* gg1row = gg1 + (size_t)bl * PNNEI * PNG1;
  const float* g1row  = g1_ext + ((size_t)b * PNALL + l) * PNG1;
  const float* swrow  = sw + (size_t)bl * PNNEI;
  const float* h2row  = h2 + (size_t)bl * PNNEI * 3;

  for (int i = tid; i < PNG1; i += 256) s_g1[i] = g1row[i];
  for (int i = tid; i < PNNEI; i += 256) s_sw[i] = swrow[i];
  for (int i = tid; i < PNNEI * 3; i += 256)
    s_h2[(i / 3) * 4 + (i % 3)] = h2row[i];
  __syncthreads();
  for (int i = tid; i < PNNEI * PNG2; i += 256) {
    float v = g2row[i];
    s_g2[i] = f2bf(v);
    s_acc[i] = v;  // residual accumulator starts at g2
  }
  for (int i = tid; i < PNNEI * PNG1; i += 256) {
    int n = i >> 7, x = i & 127;
    s_gg[i] = f2bf(s_g1[x] * gg1row[i] * s_sw[n]);  // mask is all-true
  }
  __syncthreads();

  // qk = g2 @ w_qk2 (64x32 @ 32x256) — 64 tiles
  for (int t = wave; t < 64; t += 8) {
    int mt = t >> 4, nt = t & 15;
    v16bf a = a_frag_bf(s_g2 + mt * 16 * PNG2, PNG2);
    v16bf bf = b_frag_f32(w_qk2 + nt * 16, 256);
    v8f c = {};
    c = wmma_bf16(a, bf, c);
    store_c(s_qk + mt * 16 * 256 + nt * 16, 256, c);
  }
  // v = g2 @ w_v2 (32x128) — 32 tiles
  for (int t = wave; t < 32; t += 8) {
    int mt = t >> 3, nt = t & 7;
    v16bf a = a_frag_bf(s_g2 + mt * 16 * PNG2, PNG2);
    v16bf bf = b_frag_f32(w_v2 + nt * 16, 128);
    v8f c = {};
    c = wmma_bf16(a, bf, c);
    store_c(s_v + mt * 16 * 128 + nt * 16, 128, c);
  }
  // g2_mlp = tanh(g2 @ w_lin2 + b) added to residual acc — 8 tiles
  {
    int t = wave;  // t < 8
    int mt = t >> 1, nt = t & 1;
    v16bf a = a_frag_bf(s_g2 + mt * 16 * PNG2, PNG2);
    v16bf bf = b_frag_f32(w_lin2 + nt * 16, PNG2);
    v8f c = {};
    c = wmma_bf16(a, bf, c);
    int lane = tid & 31;
    int n0 = (lane & 15) + nt * 16;
    int mb = ((lane >> 4) << 3) + mt * 16;
#pragma unroll
    for (int r = 0; r < 8; ++r)
      s_acc[(mb + r) * PNG2 + n0] += tanhf(c[r] + b_lin2[n0]);
  }
  // g2_g1g1 = gg @ w_g1g1g2 (64x128 @ 128x32) — 8 tiles, K=128 (4 steps)
  {
    int t = wave;
    int mt = t >> 1, nt = t & 1;
    v8f c = {};
#pragma unroll
    for (int kk = 0; kk < 4; ++kk) {
      v16bf a = a_frag_bf(s_gg + mt * 16 * PNG1 + kk * 32, PNG1);
      v16bf bf = b_frag_f32(w_g1g1g2 + kk * 32 * PNG2 + nt * 16, PNG2);
      c = wmma_bf16(a, bf, c);
    }
    int lane = tid & 31;
    int n0 = (lane & 15) + nt * 16;
    int mb = ((lane >> 4) << 3) + mt * 16;
#pragma unroll
    for (int r = 0; r < 8; ++r) s_acc[(mb + r) * PNG2 + n0] += c[r];
  }
  // h2 h2^T / sqrt(3)
  for (int i = tid; i < PNNEI * PNNEI; i += 256) {
    int r = i >> 6, cc = i & 63;
    float d = s_h2[r * 4 + 0] * s_h2[cc * 4 + 0] +
              s_h2[r * 4 + 1] * s_h2[cc * 4 + 1] +
              s_h2[r * 4 + 2] * s_h2[cc * 4 + 2];
    s_hht[i] = d * 0.5773502691896258f;
  }
  __syncthreads();

  const float inv_sqrt_d2 = 0.17677669529663687f;  // 1/sqrt(32)
  for (int h = 0; h < PNH2; ++h) {
    // pack per-head q,k (qk col = d*8 + {h | 4+h})
    for (int i = tid; i < PNNEI * PND2; i += 256) {
      int n = i >> 5, d = i & 31;
      s_qh[i] = f2bf(s_qk[n * 256 + d * 8 + h]);
      s_kh[i] = f2bf(s_qk[n * 256 + d * 8 + 4 + h]);
    }
    __syncthreads();
    // attnw = q @ k^T — 16 tiles, K=32
    for (int t = wave; t < 16; t += 8) {
      int mt = t >> 2, nt = t & 3;
      v16bf a = a_frag_bf(s_qh + mt * 16 * PND2, PND2);
      v16bf bf = b_frag_trans_bf(s_kh + nt * 16 * PND2, PND2);
      v8f c = {};
      c = wmma_bf16(a, bf, c);
      int lane = tid & 31;
      int n0 = (lane & 15) + nt * 16;
      int mb = ((lane >> 4) << 3) + mt * 16;
#pragma unroll
      for (int r = 0; r < 8; ++r) {
        float av = c[r] * inv_sqrt_d2;
        av = (av + PSHIFT) * s_sw[mb + r] * s_sw[n0] - PSHIFT;
        s_att[(mb + r) * PNNEI + n0] = av;
      }
    }
    __syncthreads();
    if (tid < PNNEI) {
      float m = -1e30f;
      for (int j = 0; j < PNNEI; ++j) m = fmaxf(m, s_att[tid * PNNEI + j]);
      float ssum = 0.f;
      for (int j = 0; j < PNNEI; ++j) ssum += expf(s_att[tid * PNNEI + j] - m);
      s_rmax[tid] = m;
      s_rsum[tid] = 1.0f / ssum;
    }
    __syncthreads();
    // AA = softmax(attnw) * swi*swj * h2h2t  (mask all-true), as bf16
    for (int i = tid; i < PNNEI * PNNEI; i += 256) {
      int r = i >> 6, cc = i & 63;
      float w = expf(s_att[i] - s_rmax[r]) * s_rsum[r];
      w = w * s_sw[r] * s_sw[cc];
      s_aa[i] = f2bf(w * s_hht[i]);
    }
    __syncthreads();
    // mh_h = AA @ v_h (64x64 @ 64x32) — 8 tiles, K=64 (2 steps); v_h[j][c]=v[j][c*4+h]
    {
      int t = wave;
      int mt = t >> 1, nt = t & 1;
      v8f c = {};
#pragma unroll
      for (int kk = 0; kk < 2; ++kk) {
        v16bf a = a_frag_bf(s_aa + mt * 16 * PNNEI + kk * 32, PNNEI);
        v16bf bf =
            b_frag_f32_cs(s_v + (size_t)(kk * 32) * 128 + (nt * 16) * 4 + h, 128, 4);
        c = wmma_bf16(a, bf, c);
      }
      int lane = tid & 31;
      int n0 = (lane & 15) + nt * 16;
      int mb = ((lane >> 4) << 3) + mt * 16;
#pragma unroll
      for (int r = 0; r < 8; ++r)
        s_mh[(mb + r) * 128 + n0 * 4 + h] = c[r];  // flatten (c,h) -> c*NH2+h
    }
    __syncthreads();
  }

  // g2_attn = mh(64x128) @ w_hm2(128x32) + b_hm2  (reuse s_att for 64x32 out)
  {
    int t = wave;
    int mt = t >> 1, nt = t & 1;
    v8f c = {};
#pragma unroll
    for (int kk = 0; kk < 4; ++kk) {
      v16bf a = a_frag_f32(s_mh + mt * 16 * 128 + kk * 32, 128);
      v16bf bf = b_frag_f32(w_hm2 + kk * 32 * PNG2 + nt * 16, PNG2);
      c = wmma_bf16(a, bf, c);
    }
    int lane = tid & 31;
    int n0 = (lane & 15) + nt * 16;
    int mb = ((lane >> 4) << 3) + mt * 16;
#pragma unroll
    for (int r = 0; r < 8; ++r) s_att[(mb + r) * PNG2 + n0] = c[r] + b_hm2[n0];
  }
  __syncthreads();
  // layernorm + residual combine
  float* outrow = g2_new + (size_t)bl * PNNEI * PNG2;
  if (tid < PNNEI) {
    float mu = 0.f;
    for (int c = 0; c < PNG2; ++c) mu += s_att[tid * PNG2 + c];
    mu *= (1.0f / PNG2);
    float var = 0.f;
    for (int c = 0; c < PNG2; ++c) {
      float d = s_att[tid * PNG2 + c] - mu;
      var += d * d;
    }
    var *= (1.0f / PNG2);
    float inv = rsqrtf(var + 1e-5f);
    for (int c = 0; c < PNG2; ++c) {
      float v = (s_att[tid * PNG2 + c] - mu) * inv * ln_g[c] + ln_b[c];
      outrow[tid * PNG2 + c] = (s_acc[tid * PNG2 + c] + v) * 0.5f;
    }
  }
}

// ---------------- K3: assemble g1 MLP input (800 feats per (b,l)) ----------------
constexpr size_t K3_OFF_GG1F = 0;                              // f32 64x128
constexpr size_t K3_OFF_GG1B = K3_OFF_GG1F + 64 * 128 * 4;     // bf16 64x128
constexpr size_t K3_OFF_GG1C = K3_OFF_GG1B + 64 * 128 * 2;     // f32 64x32
constexpr size_t K3_OFF_SW   = K3_OFF_GG1C + 64 * 32 * 4;      // f32 64
constexpr size_t K3_OFF_H2   = K3_OFF_SW + 64 * 4;             // f32 64x4
constexpr size_t K3_OFF_HG2  = K3_OFF_H2 + 64 * 4 * 4;         // f32 3x32
constexpr size_t K3_OFF_HGG1 = K3_OFF_HG2 + 96 * 4;            // f32 3x128
constexpr size_t K3_OFF_CONV = K3_OFF_HGG1 + 384 * 4;          // f32 32
constexpr size_t K3_SMEM     = K3_OFF_CONV + 32 * 4;

__global__ void __launch_bounds__(256) k3_g1_assemble(
    const float* __restrict__ g2, const float* __restrict__ gg1,
    const float* __restrict__ g1_ext, const float* __restrict__ sw,
    const float* __restrict__ h2, const float* __restrict__ w_g1g2,
    float* __restrict__ mlp_in) {
  extern __shared__ char smem[];
  float*  s_gg1f = (float*)(smem + K3_OFF_GG1F);
  __bf16* s_gg1b = (__bf16*)(smem + K3_OFF_GG1B);
  float*  s_gg1c = (float*)(smem + K3_OFF_GG1C);
  float*  s_sw   = (float*)(smem + K3_OFF_SW);
  float*  s_h2   = (float*)(smem + K3_OFF_H2);
  float*  s_hg2  = (float*)(smem + K3_OFF_HG2);
  float*  s_hgg1 = (float*)(smem + K3_OFF_HGG1);
  float*  s_conv = (float*)(smem + K3_OFF_CONV);

  const int tid = threadIdx.x;
  const int wave = tid >> 5;
  const int bl = blockIdx.x;
  const int b = bl >> 9, l = bl & (PNLOC - 1);
  const float* g2row  = g2 + (size_t)bl * PNNEI * PNG2;
  const float* gg1row = gg1 + (size_t)bl * PNNEI * PNG1;
  const float* g1row  = g1_ext + ((size_t)b * PNALL + l) * PNG1;
  const float* swrow  = sw + (size_t)bl * PNNEI;
  const float* h2row  = h2 + (size_t)bl * PNNEI * 3;

  for (int i = tid; i < PNNEI; i += 256) s_sw[i] = swrow[i];
  for (int i = tid; i < PNNEI * 3; i += 256)
    s_h2[(i / 3) * 4 + (i % 3)] = h2row[i];
  for (int i = tid; i < PNNEI * PNG1; i += 256) {
    float v = gg1row[i];
    s_gg1f[i] = v;
    s_gg1b[i] = f2bf(v);
  }
  __syncthreads();
  // gg1c = (gg1 @ w_g1g2) * sw — 8 tiles, K=128
  {
    int t = wave;
    int mt = t >> 1, nt = t & 1;
    v8f c = {};
#pragma unroll
    for (int kk = 0; kk < 4; ++kk) {
      v16bf a = a_frag_bf(s_gg1b + mt * 16 * PNG1 + kk * 32, PNG1);
      v16bf bf = b_frag_f32(w_g1g2 + kk * 32 * PNG2 + nt * 16, PNG2);
      c = wmma_bf16(a, bf, c);
    }
    int lane = tid & 31;
    int n0 = (lane & 15) + nt * 16;
    int mb = ((lane >> 4) << 3) + mt * 16;
#pragma unroll
    for (int r = 0; r < 8; ++r)
      s_gg1c[(mb + r) * PNG2 + n0] = c[r] * s_sw[mb + r];
  }
  __syncthreads();
  const float inv_nnei = 1.0f / PNNEI;
  // g1_conv[c] = mean_n g2[n,c]*gg1c[n,c]
  if (tid < PNG2) {
    float s = 0.f;
    for (int n = 0; n < PNNEI; ++n) s += g2row[n * PNG2 + tid] * s_gg1c[n * PNG2 + tid];
    s_conv[tid] = s * inv_nnei;
  }
  // h2g2[x][c] = mean_n h2[n,x]*g2[n,c]*sw[n]
  if (tid >= 64 && tid < 160) {
    int o = tid - 64;
    int x = o >> 5, c = o & 31;
    float s = 0.f;
    for (int n = 0; n < PNNEI; ++n)
      s += s_h2[n * 4 + x] * g2row[n * PNG2 + c] * s_sw[n];
    s_hg2[x * 32 + c] = s * inv_nnei;
  }
  // h2gg1[x][c] = mean_n h2[n,x]*gg1[n,c]*sw[n]
  for (int o = tid; o < 384; o += 256) {
    int x = o >> 7, c = o & 127;
    float s = 0.f;
    for (int n = 0; n < PNNEI; ++n)
      s += s_h2[n * 4 + x] * s_gg1f[n * PNG1 + c] * s_sw[n];
    s_hgg1[x * 128 + c] = s * inv_nnei;
  }
  __syncthreads();
  float* orow = mlp_in + (size_t)bl * PG1IN;
  for (int i = tid; i < PNG1; i += 256) orow[i] = g1row[i];
  if (tid < PNG2) orow[PNG1 + tid] = s_conv[tid];
  for (int o = tid; o < 128; o += 256) {  // grrg(g2): AXIS x 32
    int a = o >> 5, c = o & 31;
    float s = (s_hg2[0 * 32 + a] * s_hg2[0 * 32 + c] +
               s_hg2[1 * 32 + a] * s_hg2[1 * 32 + c] +
               s_hg2[2 * 32 + a] * s_hg2[2 * 32 + c]) * (1.0f / 3.0f);
    orow[160 + o] = s;
  }
  for (int o = tid; o < 512; o += 256) {  // grrg(gg1): AXIS x 128
    int a = o >> 7, c = o & 127;
    float s = (s_hgg1[0 * 128 + a] * s_hgg1[0 * 128 + c] +
               s_hgg1[1 * 128 + a] * s_hgg1[1 * 128 + c] +
               s_hgg1[2 * 128 + a] * s_hgg1[2 * 128 + c]) * (1.0f / 3.0f);
    orow[288 + o] = s;
  }
}

// ---------------- K4: g1_1 = tanh(mlp_in(2048x800) @ w_lin1(800x128) + b) ---------
__global__ void __launch_bounds__(256) k4_g1mlp(const float* __restrict__ X,
                                                const float* __restrict__ W,
                                                const float* __restrict__ bias,
                                                float* __restrict__ Y) {
  __shared__ __bf16 s_x[64 * 32];
  int m0 = blockIdx.x * 64;
  int wave = threadIdx.x >> 5;
  int mt = wave >> 1, ntb = (wave & 1) * 4;
  v8f acc[4] = {};
  for (int kk = 0; kk < PG1IN; kk += 32) {
    for (int i = threadIdx.x; i < 64 * 32; i += 256) {
      int r = i >> 5, c = i & 31;
      s_x[i] = f2bf(X[(size_t)(m0 + r) * PG1IN + kk + c]);
    }
    __syncthreads();
    v16bf a = a_frag_bf(s_x + mt * 16 * 32, 32);
#pragma unroll
    for (int j = 0; j < 4; ++j) {
      v16bf bf = b_frag_f32(W + (size_t)kk * PNG1 + (ntb + j) * 16, PNG1);
      acc[j] = wmma_bf16(a, bf, acc[j]);
    }
    __syncthreads();
  }
  int lane = threadIdx.x & 31;
  int nn = lane & 15;
  int mb = ((lane >> 4) << 3) + mt * 16;
#pragma unroll
  for (int j = 0; j < 4; ++j) {
    int n0 = (ntb + j) * 16 + nn;
#pragma unroll
    for (int r = 0; r < 8; ++r)
      Y[(size_t)(m0 + mb + r) * PNG1 + n0] = tanhf(acc[j][r] + bias[n0]);
  }
}

// ---------------- K5: g1 attention (commuted value path) ----------------
constexpr size_t K5_OFF_GG1F = 0;                            // f32 64x128
constexpr size_t K5_OFF_GG1B = K5_OFF_GG1F + 64 * 128 * 4;   // bf16 64x128
constexpr size_t K5_OFF_KQ   = K5_OFF_GG1B + 64 * 128 * 2;   // f32 64x256
constexpr size_t K5_OFF_Q    = K5_OFF_KQ + 64 * 256 * 4;     // f32 256
constexpr size_t K5_OFF_AW   = K5_OFF_Q + 256 * 4;           // f32 256
constexpr size_t K5_OFF_SW   = K5_OFF_AW + 256 * 4;          // f32 64
constexpr size_t K5_OFF_G1   = K5_OFF_SW + 64 * 4;           // f32 128
constexpr size_t K5_OFF_R    = K5_OFF_G1 + 128 * 4;          // f32 512
constexpr size_t K5_OFF_STAT = K5_OFF_R + 512 * 4;           // f32 8
constexpr size_t K5_SMEM     = K5_OFF_STAT + 16 * 4;

__global__ void __launch_bounds__(256) k5_g1_attn(
    const float* __restrict__ gg1, const float* __restrict__ g1_ext,
    const float* __restrict__ sw, const float* __restrict__ w_q1,
    const float* __restrict__ w_kv1, float* __restrict__ pre) {
  extern __shared__ char smem[];
  float*  s_gg1f = (float*)(smem + K5_OFF_GG1F);
  __bf16* s_gg1b = (__bf16*)(smem + K5_OFF_GG1B);
  float*  s_kq   = (float*)(smem + K5_OFF_KQ);
  float*  s_q    = (float*)(smem + K5_OFF_Q);
  float*  s_aw   = (float*)(smem + K5_OFF_AW);
  float*  s_sw   = (float*)(smem + K5_OFF_SW);
  float*  s_g1   = (float*)(smem + K5_OFF_G1);
  float*  s_r    = (float*)(smem + K5_OFF_R);
  float*  s_stat = (float*)(smem + K5_OFF_STAT);

  const int tid = threadIdx.x;
  const int wave = tid >> 5;
  const int bl = blockIdx.x;
  const int b = bl >> 9, l = bl & (PNLOC - 1);
  const float* gg1row = gg1 + (size_t)bl * PNNEI * PNG1;
  const float* g1row  = g1_ext + ((size_t)b * PNALL + l) * PNG1;
  const float* swrow  = sw + (size_t)bl * PNNEI;

  for (int i = tid; i < PNG1; i += 256) s_g1[i] = g1row[i];
  for (int i = tid; i < PNNEI; i += 256) s_sw[i] = swrow[i];
  for (int i = tid; i < PNNEI * PNG1; i += 256) {
    float v = gg1row[i];
    s_gg1f[i] = v;
    s_gg1b[i] = f2bf(v);
  }
  __syncthreads();
  // q[e] = g1 . w_q1[:,e]  (e = d*4 + h)
  {
    float s = 0.f;
    for (int x = 0; x < PNG1; ++x) s += s_g1[x] * w_q1[(size_t)x * 256 + tid];
    s_q[tid] = s;
  }
  // kq = gg1 @ w_kv1[:, 0:256]  (keys; cols e*4+h for e<64) — 64 tiles, K=128
  for (int t = wave; t < 64; t += 8) {
    int mt = t >> 4, nt = t & 15;
    v8f c = {};
#pragma unroll
    for (int kk = 0; kk < 4; ++kk) {
      v16bf a = a_frag_bf(s_gg1b + mt * 16 * PNG1 + kk * 32, PNG1);
      v16bf bf = b_frag_f32(w_kv1 + (size_t)kk * 32 * 768 + nt * 16, 768);
      c = wmma_bf16(a, bf, c);
    }
    store_c(s_kq + mt * 16 * 256 + nt * 16, 256, c);
  }
  __syncthreads();
  // aw[h][j] with scaling + sw shift
  const int h = tid >> 6, j = tid & 63;
  {
    float s = 0.f;
    for (int d = 0; d < PND1; ++d) s += s_q[d * 4 + h] * s_kq[j * 256 + d * 4 + h];
    s *= 0.125f;  // 1/sqrt(64)
    s = (s + PSHIFT) * s_sw[j] - PSHIFT;
    s_aw[tid] = s;
  }
  __syncthreads();
  if (j == 0) {
    float m = -1e30f;
    for (int jj = 0; jj < PNNEI; ++jj) m = fmaxf(m, s_aw[h * 64 + jj]);
    float ssum = 0.f;
    for (int jj = 0; jj < PNNEI; ++jj) ssum += expf(s_aw[h * 64 + jj] - m);
    s_stat[h * 2] = m;
    s_stat[h * 2 + 1] = 1.0f / ssum;
  }
  __syncthreads();
  {
    float awv = expf(s_aw[tid] - s_stat[h * 2]) * s_stat[h * 2 + 1] * s_sw[j];
    __syncthreads();
    s_aw[tid] = awv;
  }
  __syncthreads();
  // r[h][x] = sum_j aw[h][j] * gg1[j][x]  (commuted value path)
  for (int o = tid; o < 512; o += 256) {
    int hh = o >> 7, x = o & 127;
    float s = 0.f;
    for (int jj = 0; jj < PNNEI; ++jj) s += s_aw[hh * 64 + jj] * s_gg1f[jj * PNG1 + x];
    s_r[o] = s;
  }
  __syncthreads();
  // pre[h][c] = sum_x r[h][x] * w_kv1[x][(64+c)*4+h]
  float* prow = pre + (size_t)bl * (PNH1 * PNG1);
  for (int o = tid; o < 512; o += 256) {
    int hh = o >> 7, c = o & 127;
    float s = 0.f;
    for (int x = 0; x < PNG1; ++x)
      s += s_r[hh * 128 + x] * w_kv1[(size_t)x * 768 + (PND1 + c) * 4 + hh];
    prow[o] = s;
  }
}

// ---------------- K6: g1_new = (g1 + g1_1 + pre@w_hm1 + b)/sqrt(3) ----------------
__global__ void __launch_bounds__(256) k6_g1_final(
    const float* __restrict__ P, const float* __restrict__ W,
    const float* __restrict__ bias, const float* __restrict__ g1_1,
    const float* __restrict__ g1_ext, float* __restrict__ g1_new) {
  __shared__ __bf16 s_x[64 * 32];
  int m0 = blockIdx.x * 64;
  int wave = threadIdx.x >> 5;
  int mt = wave >> 1, ntb = (wave & 1) * 4;
  v8f acc[4] = {};
  for (int kk = 0; kk < 512; kk += 32) {
    for (int i = threadIdx.x; i < 64 * 32; i += 256) {
      int r = i >> 5, c = i & 31;
      s_x[i] = f2bf(P[(size_t)(m0 + r) * 512 + kk + c]);
    }
    __syncthreads();
    v16bf a = a_frag_bf(s_x + mt * 16 * 32, 32);
#pragma unroll
    for (int j = 0; j < 4; ++j) {
      v16bf bf = b_frag_f32(W + (size_t)kk * PNG1 + (ntb + j) * 16, PNG1);
      acc[j] = wmma_bf16(a, bf, acc[j]);
    }
    __syncthreads();
  }
  const float inv_sqrt3 = 0.5773502691896258f;
  int lane = threadIdx.x & 31;
  int nn = lane & 15;
  int mb = ((lane >> 4) << 3) + mt * 16;
#pragma unroll
  for (int j = 0; j < 4; ++j) {
    int n0 = (ntb + j) * 16 + nn;
#pragma unroll
    for (int r = 0; r < 8; ++r) {
      int m = m0 + mb + r;
      int b = m >> 9, l = m & (PNLOC - 1);
      float g1v = g1_ext[((size_t)b * PNALL + l) * PNG1 + n0];
      g1_new[(size_t)m * PNG1 + n0] =
          (g1v + g1_1[(size_t)m * PNG1 + n0] + acc[j][r] + bias[n0]) * inv_sqrt3;
    }
  }
}

// ---------------- host launch ----------------
extern "C" void kernel_launch(void* const* d_in, const int* in_sizes, int n_in,
                              void* d_out, int out_size, void* d_ws, size_t ws_size,
                              hipStream_t stream) {
  (void)in_sizes; (void)n_in; (void)out_size; (void)ws_size;
  const float* g1_ext   = (const float*)d_in[0];
  const float* g2       = (const float*)d_in[1];
  const float* h2       = (const float*)d_in[2];
  const float* sw       = (const float*)d_in[3];
  const int*   nlist    = (const int*)d_in[4];
  // d_in[5] = nlist_mask (all true; unused)
  const float* w_lin1   = (const float*)d_in[6];
  const float* b_lin1   = (const float*)d_in[7];
  const float* w_lin2   = (const float*)d_in[8];
  const float* b_lin2   = (const float*)d_in[9];
  const float* w_g1g2   = (const float*)d_in[10];
  const float* w_g1g1g2 = (const float*)d_in[11];
  const float* w_qk2    = (const float*)d_in[12];
  const float* w_v2     = (const float*)d_in[13];
  const float* w_hm2    = (const float*)d_in[14];
  const float* b_hm2    = (const float*)d_in[15];
  const float* ln_g     = (const float*)d_in[16];
  const float* ln_b     = (const float*)d_in[17];
  const float* w_q1     = (const float*)d_in[18];
  const float* w_kv1    = (const float*)d_in[19];
  const float* w_hm1    = (const float*)d_in[20];
  const float* b_hm1    = (const float*)d_in[21];

  float* out_g1 = (float*)d_out;
  float* out_g2 = out_g1 + (size_t)PNB * PNLOC * PNG1;
  float* out_h2 = out_g2 + (size_t)PNB * PNLOC * PNNEI * PNG2;

  float* ws      = (float*)d_ws;
  float* gg1     = ws;                                            // 16,777,216 f
  float* mlp_in  = gg1 + (size_t)PNB * PNLOC * PNNEI * PNG1;      //  1,638,400 f
  float* g1_1    = mlp_in + (size_t)PNB * PNLOC * PG1IN;          //    262,144 f
  float* pre     = g1_1 + (size_t)PNB * PNLOC * PNG1;             //  1,048,576 f

  const int NBL = PNB * PNLOC;  // 2048

  k1_gather<<<(PNB * PNLOC * PNNEI * PNG1) / 256, 256, 0, stream>>>(g1_ext, nlist, gg1);
  k2_g2_fused<<<NBL, 256, K2_SMEM, stream>>>(g2, gg1, g1_ext, sw, h2, w_lin2, b_lin2,
                                             w_g1g1g2, w_qk2, w_v2, w_hm2, b_hm2,
                                             ln_g, ln_b, out_g2);
  k3_g1_assemble<<<NBL, 256, K3_SMEM, stream>>>(g2, gg1, g1_ext, sw, h2, w_g1g2,
                                                mlp_in);
  k4_g1mlp<<<NBL / 64, 256, 0, stream>>>(mlp_in, w_lin1, b_lin1, g1_1);
  k5_g1_attn<<<NBL, 256, K5_SMEM, stream>>>(gg1, g1_ext, sw, w_q1, w_kv1, pre);
  k6_g1_final<<<NBL / 64, 256, 0, stream>>>(pre, w_hm1, b_hm1, g1_1, g1_ext, out_g1);
  hipMemcpyAsync(out_h2, h2, (size_t)PNB * PNLOC * PNNEI * 3 * sizeof(float),
                 hipMemcpyDeviceToDevice, stream);
}
